// DynamicCNN4_78245714198704
// MI455X (gfx1250) — matrix operand order repaired and checked by reading, not compiled, not run on hardware
//
#include <hip/hip_runtime.h>
#include <hip/hip_bf16.h>

// ---------------------------------------------------------------------------
// DynamicCNN4 for MI455X (gfx1250, wave32).
// Phase 1: hypernet MLP as WMMA f16->f32 GEMM chain -> per-sample conv weights
//          (stored to d_ws, stride 816, 16384*816*4 = 53.5 MB, L2-resident).
// Phase 2: wave-per-sample fused conv/relu/maxpool chain in VALU f32 with
//          LDS-staged activations (staged via async global->LDS copies);
//          adds residual state and writes d_out.
// ---------------------------------------------------------------------------

typedef _Float16 v16h __attribute__((ext_vector_type(16)));
typedef float    v8f  __attribute__((ext_vector_type(8)));

#define NB      16384     // batch
#define WSTRIDE 816       // padded row stride of generated weight matrix
#define WTOT    811

// ---- weight-vector chunk offsets (per sample row of w) --------------------
// w1: [0,27)   co*9 + dy*3 + dx                     (cout=3, cin=1)
// b1: [27,30)
// w2: [30,192) co*27 + dy*9 + dx*3 + ci             (cout=6, cin=3)
// b2: [192,198)
// w3: [198,360) co*54 + dy*18 + dx*6 + ci           (cout=3, cin=6)
// b3: [360,363)
// w4: [363,795) co*27 + dy*9 + dx*3 + ci            (cout=16, cin=3)
// b4: [795,811)

// Async copy of one 32-bit element from global memory directly into LDS.
// dsaddr = low 32 bits of the generic shared-memory pointer (ISA: LDS aperture
// address is addr[31:0]); tracked with ASYNCcnt.
__device__ __forceinline__ void async_g2l_b32(void* lds_dst, const void* gsrc) {
    unsigned dst           = (unsigned)(unsigned long long)lds_dst;
    unsigned long long src = (unsigned long long)gsrc;
    asm volatile("global_load_async_to_lds_b32 %0, %1, off"
                 :: "v"(dst), "v"(src) : "memory");
}

__device__ __forceinline__ void wait_async_all() {
    asm volatile("s_wait_asynccnt 0x0" ::: "memory");
}

// ===========================================================================
// Phase 1: hypernet.  grid = 128 blocks x 256 thr (8 waves); wave = 16 rows.
// ===========================================================================
__global__ __launch_bounds__(256)
void hyper_kernel(const float* __restrict__ state,
                  const float* __restrict__ W1, const float* __restrict__ b1,
                  const float* __restrict__ W2, const float* __restrict__ b2,
                  const float* __restrict__ W3, const float* __restrict__ b3,
                  float* __restrict__ wout)
{
    __shared__ float ht[8][16][16];           // per-wave 16x16 staging tile

    const int wave = threadIdx.x >> 5;
    const int lane = threadIdx.x & 31;
    const int hi   = lane >> 4;               // 0: lanes 0-15, 1: lanes 16-31
    const int mr   = lane & 15;               // row (A) / col (B, C) in tile
    const int m0   = (blockIdx.x * 8 + wave) * 16;

    // ---- A fragment from state tile (16-bit A 16x32 layout, K padded) ----
    v16h a = {};
    #pragma unroll
    for (int e = 0; e < 16; ++e)
        if (!hi) a[e] = (_Float16)state[(m0 + mr) * 16 + e];

    // ---- B fragment for W1 (B rows striped across lanes; K>=16 zero) -----
    v16h bw = {};
    if (!hi) {
        #pragma unroll
        for (int e = 0; e < 16; ++e) bw[e] = (_Float16)W1[e * 16 + mr];
    }

    v8f c = {};
    c = __builtin_amdgcn_wmma_f32_16x16x32_f16(false, a, false, bw,
                                               (short)0, c, false, false);
    float bias = b1[mr];
    #pragma unroll
    for (int r = 0; r < 8; ++r) c[r] = fmaxf(c[r] + bias, 0.f);

    // ---- round-trip through LDS: C layout -> A layout --------------------
    #pragma unroll
    for (int r = 0; r < 8; ++r) ht[wave][hi * 8 + r][mr] = c[r];
    __syncthreads();
    v16h ah = {};
    #pragma unroll
    for (int e = 0; e < 16; ++e) if (!hi) ah[e] = (_Float16)ht[wave][mr][e];

    v16h bw2 = {};
    if (!hi) {
        #pragma unroll
        for (int e = 0; e < 16; ++e) bw2[e] = (_Float16)W2[e * 16 + mr];
    }
    v8f c2 = {};
    c2 = __builtin_amdgcn_wmma_f32_16x16x32_f16(false, ah, false, bw2,
                                                (short)0, c2, false, false);
    bias = b2[mr];
    #pragma unroll
    for (int r = 0; r < 8; ++r) c2[r] = fmaxf(c2[r] + bias, 0.f);

    __syncthreads();
    #pragma unroll
    for (int r = 0; r < 8; ++r) ht[wave][hi * 8 + r][mr] = c2[r];
    __syncthreads();
    v16h ah2 = {};
    #pragma unroll
    for (int e = 0; e < 16; ++e) if (!hi) ah2[e] = (_Float16)ht[wave][mr][e];

    // ---- big GEMM: h2[16x16] @ W3[16x811], 51 N-tiles --------------------
    for (int n0 = 0; n0 < WTOT; n0 += 16) {
        const int n = n0 + mr;
        v16h bw3 = {};
        if (!hi && n < WTOT) {
            #pragma unroll
            for (int e = 0; e < 16; ++e) bw3[e] = (_Float16)W3[e * WTOT + n];
        }
        v8f d = {};
        d = __builtin_amdgcn_wmma_f32_16x16x32_f16(false, ah2, false, bw3,
                                                   (short)0, d, false, false);
        const float b3v = (n < WTOT) ? b3[n] : 0.f;
        if (n < WTOT) {
            #pragma unroll
            for (int r = 0; r < 8; ++r) {
                const int row = m0 + hi * 8 + r;
                wout[row * WSTRIDE + n] = d[r] + b3v;
            }
        }
    }
}

// ===========================================================================
// Phase 2: per-sample conv chain.  One wave32 per sample, 4 waves per block.
// ===========================================================================
#define WPB 4

__global__ __launch_bounds__(32 * WPB)
void conv_kernel(const float* __restrict__ image,   // [28,28,B] channel-last
                 const float* __restrict__ state,   // [B,16]
                 const float* __restrict__ wvecs,   // [B, WSTRIDE]
                 float* __restrict__ out)           // [B,16]
{
    __shared__ float s_img[WPB][784];
    __shared__ float s_w  [WPB][WSTRIDE];
    __shared__ float s_a1 [WPB][507];   // 13x13x3 after conv1+relu+pool
    __shared__ float s_a2 [WPB][150];   // 5x5x6   after conv2+relu+pool
    __shared__ float s_a3 [WPB][32];    // 3x3x3   after conv3+relu

    const int wave   = threadIdx.x >> 5;
    const int lane   = threadIdx.x & 31;
    const int sample = blockIdx.x * WPB + wave;

    float*       img = s_img[wave];
    float*       w   = s_w[wave];
    float*       a1  = s_a1[wave];
    float*       a2  = s_a2[wave];
    float*       a3  = s_a3[wave];

    // Async-stage weight row (coalesced) and image column (strided; whole
    // 51 MB image is L2-resident) straight into LDS, bypassing VGPRs.
    for (int i = lane; i < WTOT; i += 32)
        async_g2l_b32(&w[i], wvecs + (size_t)sample * WSTRIDE + i);
    for (int i = lane; i < 784; i += 32)
        async_g2l_b32(&img[i], image + (size_t)i * NB + sample);
    wait_async_all();
    __syncthreads();

    // ---- conv1 (1->3) + relu + pool2: 28x28 -> 13x13x3 -------------------
    float k1[27], bb1[3];
    #pragma unroll
    for (int i = 0; i < 27; ++i) k1[i] = w[i];
    #pragma unroll
    for (int i = 0; i < 3; ++i) bb1[i] = w[27 + i];

    for (int o = lane; o < 507; o += 32) {
        const int c  = o % 3;
        const int xy = o / 3;
        const int x  = (xy % 13) * 2;
        const int y  = (xy / 13) * 2;
        float m = 0.f;                       // relu then max == max(0, max4)
        #pragma unroll
        for (int py = 0; py < 2; ++py)
            #pragma unroll
            for (int px = 0; px < 2; ++px) {
                float acc = bb1[c];
                #pragma unroll
                for (int dy = 0; dy < 3; ++dy)
                    #pragma unroll
                    for (int dx = 0; dx < 3; ++dx)
                        acc = fmaf(img[(y + py + dy) * 28 + (x + px + dx)],
                                   k1[c * 9 + dy * 3 + dx], acc);
                m = fmaxf(m, acc);
            }
        a1[o] = m;
    }
    __syncthreads();

    // ---- conv2 (3->6) + relu + pool2: 13x13x3 -> 5x5x6 -------------------
    for (int o = lane; o < 150; o += 32) {
        const int co = o % 6;
        const int xy = o / 6;
        const int x  = (xy % 5) * 2;
        const int y  = (xy / 5) * 2;
        const float bb = w[192 + co];
        float m = 0.f;
        #pragma unroll
        for (int py = 0; py < 2; ++py)
            #pragma unroll
            for (int px = 0; px < 2; ++px) {
                float acc = bb;
                #pragma unroll
                for (int dy = 0; dy < 3; ++dy)
                    #pragma unroll
                    for (int dx = 0; dx < 3; ++dx)
                        #pragma unroll
                        for (int ci = 0; ci < 3; ++ci)
                            acc = fmaf(a1[((y + py + dy) * 13 + (x + px + dx)) * 3 + ci],
                                       w[30 + co * 27 + dy * 9 + dx * 3 + ci], acc);
                m = fmaxf(m, acc);
            }
        a2[o] = m;
    }
    __syncthreads();

    // ---- conv3 (6->3) + relu: 5x5x6 -> 3x3x3 -----------------------------
    if (lane < 27) {
        const int co = lane % 3;
        const int xy = lane / 3;
        const int x  = xy % 3;
        const int y  = xy / 3;
        float acc = w[360 + co];
        #pragma unroll
        for (int dy = 0; dy < 3; ++dy)
            #pragma unroll
            for (int dx = 0; dx < 3; ++dx)
                #pragma unroll
                for (int ci = 0; ci < 6; ++ci)
                    acc = fmaf(a2[((y + dy) * 5 + (x + dx)) * 6 + ci],
                               w[198 + co * 54 + dy * 18 + dx * 6 + ci], acc);
        a3[lane] = fmaxf(acc, 0.f);
    }
    __syncthreads();

    // ---- conv4 (3->16): 3x3x3 -> 16, add residual state ------------------
    if (lane < 16) {
        float acc = w[795 + lane];
        #pragma unroll
        for (int dy = 0; dy < 3; ++dy)
            #pragma unroll
            for (int dx = 0; dx < 3; ++dx)
                #pragma unroll
                for (int ci = 0; ci < 3; ++ci)
                    acc = fmaf(a3[(dy * 3 + dx) * 3 + ci],
                               w[363 + lane * 27 + dy * 9 + dx * 3 + ci], acc);
        out[sample * 16 + lane] = state[sample * 16 + lane] + acc;
    }
}

// ===========================================================================
extern "C" void kernel_launch(void* const* d_in, const int* in_sizes, int n_in,
                              void* d_out, int out_size, void* d_ws, size_t ws_size,
                              hipStream_t stream) {
    const float* image = (const float*)d_in[0];   // [1,28,28,16384]
    const float* state = (const float*)d_in[1];   // [16384,16]
    const float* W1    = (const float*)d_in[2];   // [16,16]
    const float* b1    = (const float*)d_in[3];   // [16]
    const float* W2    = (const float*)d_in[4];   // [16,16]
    const float* b2    = (const float*)d_in[5];   // [16]
    const float* W3    = (const float*)d_in[6];   // [16,811]
    const float* b3    = (const float*)d_in[7];   // [811]
    float* outp = (float*)d_out;
    float* wws  = (float*)d_ws;                   // needs 16384*816*4 = 53.5 MB

    // 1024 M-tiles, 8 waves/block -> 128 blocks
    hyper_kernel<<<NB / 16 / 8, 256, 0, stream>>>(state, W1, b1, W2, b2, W3, b3, wws);
    // one wave per sample, 4 waves/block -> 4096 blocks
    conv_kernel<<<NB / WPB, 32 * WPB, 0, stream>>>(image, state, wws, outp);
}
